// MultiHeadAttention_21818433864071
// MI455X (gfx1250) — matrix operand order repaired and checked by reading, not compiled, not run on hardware
//
#include <hip/hip_runtime.h>

// MI455X / gfx1250, wave32, WMMA f32<-f16 path.
// MHA forward: QKV proj (WMMA GEMM) -> flash attention (WMMA) -> out proj (WMMA GEMM).
// Round 2: K-step 64, double-buffered LDS (1 barrier/iter), padded V-transpose stride.

typedef __attribute__((ext_vector_type(16))) _Float16 v16h;
typedef __attribute__((ext_vector_type(4)))  _Float16 v4h;
typedef __attribute__((ext_vector_type(8)))  float    v8f;
typedef __attribute__((ext_vector_type(4)))  float    v4f;

#define BSZ    4
#define SEQ    2048
#define DMODEL 1024
#define NHEAD  16
#define DKH    64
#define NEGV   (-1000000000.0f)
#define SCALE  0.125f   // 1/sqrt(64)
#define VSTR   72       // padded LDS row stride for transposed V (16B aligned, bank-spread)

__device__ __forceinline__ v8f wmma_f16(v16h a, v16h b, v8f c) {
  return __builtin_amdgcn_wmma_f32_16x16x32_f16(
      /*neg_a=*/false, a, /*neg_b=*/false, b,
      /*c_mod=*/(short)0, c, /*reuse_a=*/false, /*reuse_b=*/false);
}

// B-fragment (32 x 16, 16-bit): lane = col (l%16), 16 contiguous K halves at +16*(l/16).
__device__ __forceinline__ v16h lds_frag_contig(const _Float16* p) {
  union { v4f f[2]; v16h h; } u;
  u.f[0] = *(const v4f*)(p);
  u.f[1] = *(const v4f*)(p + 8);
  return u.h;
}

// A-fragment (16 x 32, 16-bit): lane = row (l%16); halves 0-15 = K[16g..16g+15].
__device__ __forceinline__ v16h lds_frag_a(const _Float16* row, int g) {
  union { v4f f[2]; v16h h; } u;
  u.f[0] = *(const v4f*)(row + 16 * g);
  u.f[1] = *(const v4f*)(row + 16 * g + 8);
  return u.h;
}

// ---------------- weight convert f32 -> f16 ----------------
__global__ void wconv_kernel(const float* __restrict__ s, _Float16* __restrict__ d, int n) {
  int i = (blockIdx.x * blockDim.x + threadIdx.x) * 4;
  if (i + 3 < n) {
    v4f x = *(const v4f*)(s + i);
    v4h h = {(_Float16)x.x, (_Float16)x.y, (_Float16)x.z, (_Float16)x.w};
    *(v4h*)(d + i) = h;
  }
}

// ---------------- QKV projection: Y = X @ W^T + b, head-split f16 out ----------------
// block 256 (8 waves), tile 128(M) x 64(N), K step 64, double-buffered LDS.
__global__ __launch_bounds__(256) void qkv_proj_kernel(
    const float* __restrict__ X, const _Float16* __restrict__ W,
    const float* __restrict__ bias, _Float16* __restrict__ Yh) {
  __shared__ __align__(16) _Float16 As[2][128 * 64];
  __shared__ __align__(16) _Float16 Bs[2][64 * 64];

  const int tid = threadIdx.x, lane = tid & 31, wid = tid >> 5;
  const int wm = wid & 3, wn = wid >> 2;
  const int g = lane >> 4, ln = lane & 15;
  const int m0 = blockIdx.x * 128, n0 = blockIdx.y * 64;

  v8f acc[2][2];
#pragma unroll
  for (int i = 0; i < 2; ++i)
#pragma unroll
    for (int j = 0; j < 2; ++j) acc[i][j] = (v8f){0.f,0.f,0.f,0.f,0.f,0.f,0.f,0.f};

  for (int t = 0; t < DMODEL / 64; ++t) {
    const int k0 = t * 64, buf = t & 1;
    // A tile 128x64: f32 global -> f16 LDS  (2048 chunks of 4 floats)
#pragma unroll
    for (int e = 0; e < 8; ++e) {
      int c = tid + 256 * e;
      int row = c >> 4, col = (c & 15) << 2;
      v4f x = *(const v4f*)(X + (size_t)(m0 + row) * DMODEL + k0 + col);
      v4h h = {(_Float16)x.x, (_Float16)x.y, (_Float16)x.z, (_Float16)x.w};
      *(v4h*)(&As[buf][row * 64 + col]) = h;
    }
    // B tile 64x64 from W(f16)  (512 chunks of 8 halves)
#pragma unroll
    for (int e = 0; e < 2; ++e) {
      int c = tid + 256 * e;
      int row = c >> 3, col = (c & 7) << 3;
      *(v4f*)(&Bs[buf][row * 64 + col]) =
          *(const v4f*)(W + (size_t)(n0 + row) * DMODEL + k0 + col);
    }
    __syncthreads();   // single barrier per iter: next iter writes the other buffer

#pragma unroll
    for (int kk = 0; kk < 2; ++kk) {
      v16h af[2], bf[2];
#pragma unroll
      for (int i = 0; i < 2; ++i)
        af[i] = lds_frag_a(&As[buf][(wm * 32 + 16 * i + ln) * 64 + 32 * kk], g);
#pragma unroll
      for (int j = 0; j < 2; ++j)
        bf[j] = lds_frag_contig(&Bs[buf][(wn * 32 + 16 * j + ln) * 64 + 32 * kk + 16 * g]);
#pragma unroll
      for (int i = 0; i < 2; ++i)
#pragma unroll
        for (int j = 0; j < 2; ++j) acc[i][j] = wmma_f16(af[i], bf[j], acc[i][j]);
    }
  }

  // epilogue: + bias, write f16 head-split [B, H, S, DKH]
#pragma unroll
  for (int j = 0; j < 2; ++j) {
    int ng = n0 + wn * 32 + 16 * j + ln;
    float bj = bias[ng];
    int hh = ng >> 6, dk = ng & 63;
#pragma unroll
    for (int i = 0; i < 2; ++i)
#pragma unroll
      for (int r = 0; r < 8; ++r) {
        int mg = m0 + wm * 32 + 16 * i + r + 8 * g;
        int b = mg >> 11, s = mg & (SEQ - 1);
        float v = acc[i][j][r] + bj;
        Yh[(((size_t)(b * NHEAD + hh) * SEQ) + s) * DKH + dk] = (_Float16)v;
      }
  }
}

// ---------------- flash attention per (bh, 64-row q tile) ----------------
// block 128 (4 waves); wave owns 16 q rows; key tiles of 64; double-buffered K/V.
__global__ __launch_bounds__(128) void attn_kernel(
    const _Float16* __restrict__ Qh, const _Float16* __restrict__ Kh,
    const _Float16* __restrict__ Vh, const int* __restrict__ mask,
    _Float16* __restrict__ Yc) {
  __shared__ __align__(16) _Float16 Qs[64 * 64];
  __shared__ __align__(16) _Float16 Ks[2][64 * 64];      // [key][dk]
  __shared__ __align__(16) _Float16 Vs[2][64 * VSTR];    // transposed: [dk][key], padded
  __shared__ __align__(16) _Float16 Ps[4 * 16 * 64];     // per-wave P patch
  __shared__ int kms[2][64];

  const int tid = threadIdx.x, lane = tid & 31, wid = tid >> 5;
  const int g = lane >> 4, ln = lane & 15;
  const int q0 = blockIdx.x * 64;
  const int bh = blockIdx.y, b = bh >> 4, h = bh & 15;

  const _Float16* Qg = Qh + ((size_t)bh * SEQ + q0) * DKH;
  const _Float16* Kg = Kh + (size_t)bh * SEQ * DKH;
  const _Float16* Vg = Vh + (size_t)bh * SEQ * DKH;
  const int* mb = mask + b * SEQ;

  // stage Q tile (contiguous 4096 halves)
#pragma unroll
  for (int e = 0; e < 4; ++e) {
    int c = tid + 128 * e;
    *(v4f*)(Qs + c * 8) = *(const v4f*)(Qg + c * 8);
  }
  __syncthreads();

  v16h qf[2];
#pragma unroll
  for (int kk = 0; kk < 2; ++kk)
    qf[kk] = lds_frag_a(Qs + (wid * 16 + ln) * 64 + 32 * kk, g);

  v8f O[4];
  float M[8], L[8];
#pragma unroll
  for (int d = 0; d < 4; ++d) O[d] = (v8f){0.f,0.f,0.f,0.f,0.f,0.f,0.f,0.f};
#pragma unroll
  for (int r = 0; r < 8; ++r) { M[r] = -1e30f; L[r] = 0.f; }

  for (int t = 0; t < SEQ / 64; ++t) {
    const int kt = t * 64, buf = t & 1;
    // stage K tile (contiguous)
#pragma unroll
    for (int e = 0; e < 4; ++e) {
      int c = tid + 128 * e;
      *(v4f*)(&Ks[buf][c * 8]) = *(const v4f*)(Kg + (size_t)kt * DKH + c * 8);
    }
    // stage V tile transposed -> Vs[dk][key] (padded row stride)
    for (int e = 0; e < 32; ++e) {
      int i = tid + 128 * e;
      int key = i >> 6, dk = i & 63;
      Vs[buf][dk * VSTR + key] = Vg[(size_t)(kt + key) * DKH + dk];
    }
    if (tid < 64) kms[buf][tid] = mb[kt + tid];
    if (t + 1 < SEQ / 64) {
      __builtin_prefetch(Kg + (size_t)(kt + 64) * DKH, 0, 0);
      __builtin_prefetch(Vg + (size_t)(kt + 64) * DKH, 0, 0);
    }
    __syncthreads();   // single barrier: next iter stages the other buffer

    int km[4];
#pragma unroll
    for (int j = 0; j < 4; ++j) km[j] = kms[buf][ln + 16 * j];

    // scores: 16x64 per wave, 8 WMMAs
    v8f sc[4];
#pragma unroll
    for (int j = 0; j < 4; ++j) {
      v8f c = (v8f){0.f,0.f,0.f,0.f,0.f,0.f,0.f,0.f};
#pragma unroll
      for (int kk = 0; kk < 2; ++kk) {
        v16h kf = lds_frag_contig(&Ks[buf][(ln + 16 * j) * 64 + 32 * kk + 16 * g]);
        c = wmma_f16(qf[kk], kf, c);
      }
      sc[j] = c;
    }

    // online softmax (rows r+8g, 16-lane N groups)
    float p[4][8];
#pragma unroll
    for (int r = 0; r < 8; ++r) {
      float s[4];
#pragma unroll
      for (int j = 0; j < 4; ++j) s[j] = km[j] ? sc[j][r] * SCALE : NEGV;
      float mx = fmaxf(fmaxf(s[0], s[1]), fmaxf(s[2], s[3]));
#pragma unroll
      for (int off = 1; off < 16; off <<= 1) mx = fmaxf(mx, __shfl_xor(mx, off, 16));
      float nm = fmaxf(M[r], mx);
      float alpha = __expf(M[r] - nm);
      float sum = 0.f;
#pragma unroll
      for (int j = 0; j < 4; ++j) {
        float pv = km[j] ? __expf(s[j] - nm) : 0.f;  // exact zero for masked keys
        p[j][r] = pv; sum += pv;
      }
#pragma unroll
      for (int off = 1; off < 16; off <<= 1) sum += __shfl_xor(sum, off, 16);
      M[r] = nm;
      L[r] = L[r] * alpha + sum;
#pragma unroll
      for (int d = 0; d < 4; ++d) O[d][r] *= alpha;
    }

    // P (C layout) -> per-wave LDS -> A fragments
    _Float16* Pw = Ps + wid * (16 * 64);
#pragma unroll
    for (int j = 0; j < 4; ++j)
#pragma unroll
      for (int r = 0; r < 8; ++r)
        Pw[(r + 8 * g) * 64 + ln + 16 * j] = (_Float16)p[j][r];

    v16h pf[2];
#pragma unroll
    for (int kk = 0; kk < 2; ++kk) pf[kk] = lds_frag_a(Pw + ln * 64 + 32 * kk, g);

    // O += P @ V, 8 WMMAs
#pragma unroll
    for (int d = 0; d < 4; ++d)
#pragma unroll
      for (int kk = 0; kk < 2; ++kk) {
        v16h vf = lds_frag_contig(&Vs[buf][(ln + 16 * d) * VSTR + 32 * kk + 16 * g]);
        O[d] = wmma_f16(pf[kk], vf, O[d]);
      }
  }

  // normalize, apply query mask, write concat layout [B, S, H*DKH] as f16
  float rinv[8];
#pragma unroll
  for (int r = 0; r < 8; ++r) {
    int qm = mb[q0 + wid * 16 + r + 8 * g];
    rinv[r] = (qm != 0 && L[r] > 0.f) ? 1.f / L[r] : 0.f;
  }
  _Float16* Yo = Yc + (size_t)b * SEQ * DMODEL + (size_t)h * DKH;
#pragma unroll
  for (int d = 0; d < 4; ++d)
#pragma unroll
    for (int r = 0; r < 8; ++r) {
      int srow = q0 + wid * 16 + r + 8 * g;
      Yo[(size_t)srow * DMODEL + ln + 16 * d] = (_Float16)(O[d][r] * rinv[r]);
    }
}

// ---------------- output projection: Out = Yc @ Wo^T + bo (f32 out) ----------------
__global__ __launch_bounds__(256) void out_proj_kernel(
    const _Float16* __restrict__ Yc, const _Float16* __restrict__ W,
    const float* __restrict__ bias, float* __restrict__ Out) {
  __shared__ __align__(16) _Float16 As[2][128 * 64];
  __shared__ __align__(16) _Float16 Bs[2][64 * 64];

  const int tid = threadIdx.x, lane = tid & 31, wid = tid >> 5;
  const int wm = wid & 3, wn = wid >> 2;
  const int g = lane >> 4, ln = lane & 15;
  const int m0 = blockIdx.x * 128, n0 = blockIdx.y * 64;

  v8f acc[2][2];
#pragma unroll
  for (int i = 0; i < 2; ++i)
#pragma unroll
    for (int j = 0; j < 2; ++j) acc[i][j] = (v8f){0.f,0.f,0.f,0.f,0.f,0.f,0.f,0.f};

  for (int t = 0; t < DMODEL / 64; ++t) {
    const int k0 = t * 64, buf = t & 1;
    // A tile 128x64 halves (1024 chunks of 8 halves)
#pragma unroll
    for (int e = 0; e < 4; ++e) {
      int c = tid + 256 * e;
      int row = c >> 3, col = (c & 7) << 3;
      *(v4f*)(&As[buf][row * 64 + col]) =
          *(const v4f*)(Yc + (size_t)(m0 + row) * DMODEL + k0 + col);
    }
    // B tile 64x64 halves (512 chunks of 8 halves)
#pragma unroll
    for (int e = 0; e < 2; ++e) {
      int c = tid + 256 * e;
      int row = c >> 3, col = (c & 7) << 3;
      *(v4f*)(&Bs[buf][row * 64 + col]) =
          *(const v4f*)(W + (size_t)(n0 + row) * DMODEL + k0 + col);
    }
    __syncthreads();

#pragma unroll
    for (int kk = 0; kk < 2; ++kk) {
      v16h af[2], bf[2];
#pragma unroll
      for (int i = 0; i < 2; ++i)
        af[i] = lds_frag_a(&As[buf][(wm * 32 + 16 * i + ln) * 64 + 32 * kk], g);
#pragma unroll
      for (int j = 0; j < 2; ++j)
        bf[j] = lds_frag_contig(&Bs[buf][(wn * 32 + 16 * j + ln) * 64 + 32 * kk + 16 * g]);
#pragma unroll
      for (int i = 0; i < 2; ++i)
#pragma unroll
        for (int j = 0; j < 2; ++j) acc[i][j] = wmma_f16(af[i], bf[j], acc[i][j]);
    }
  }

#pragma unroll
  for (int j = 0; j < 2; ++j) {
    int ng = n0 + wn * 32 + 16 * j + ln;
    float bj = bias[ng];
#pragma unroll
    for (int i = 0; i < 2; ++i)
#pragma unroll
      for (int r = 0; r < 8; ++r) {
        int mg = m0 + wm * 32 + 16 * i + r + 8 * g;
        Out[(size_t)mg * DMODEL + ng] = acc[i][j][r] + bj;
      }
  }
}

extern "C" void kernel_launch(void* const* d_in, const int* in_sizes, int n_in,
                              void* d_out, int out_size, void* d_ws, size_t ws_size,
                              hipStream_t stream) {
  (void)in_sizes; (void)n_in; (void)out_size; (void)ws_size;
  const float* q    = (const float*)d_in[0];
  const float* k    = (const float*)d_in[1];
  const float* v    = (const float*)d_in[2];
  const int*   mask = (const int*)d_in[3];
  const float* Wq   = (const float*)d_in[4];
  const float* bq   = (const float*)d_in[5];
  const float* Wk   = (const float*)d_in[6];
  const float* bk   = (const float*)d_in[7];
  const float* Wv   = (const float*)d_in[8];
  const float* bv   = (const float*)d_in[9];
  const float* Wo   = (const float*)d_in[10];
  const float* bo   = (const float*)d_in[11];
  float* out = (float*)d_out;

  // workspace carve-up (~72 MB): 4x f16 weights (2MB) + Qh/Kh/Vh/Yc (16MB each)
  char* w = (char*)d_ws;
  const size_t WB = (size_t)DMODEL * DMODEL * sizeof(_Float16);
  const size_t TB = (size_t)BSZ * SEQ * DMODEL * sizeof(_Float16);
  _Float16* Wq16 = (_Float16*)w; w += WB;
  _Float16* Wk16 = (_Float16*)w; w += WB;
  _Float16* Wv16 = (_Float16*)w; w += WB;
  _Float16* Wo16 = (_Float16*)w; w += WB;
  _Float16* Qh   = (_Float16*)w; w += TB;
  _Float16* Kh   = (_Float16*)w; w += TB;
  _Float16* Vh   = (_Float16*)w; w += TB;
  _Float16* Yc   = (_Float16*)w; w += TB;

  const int nW = DMODEL * DMODEL;
  wconv_kernel<<<nW / 1024, 256, 0, stream>>>(Wq, Wq16, nW);
  wconv_kernel<<<nW / 1024, 256, 0, stream>>>(Wk, Wk16, nW);
  wconv_kernel<<<nW / 1024, 256, 0, stream>>>(Wv, Wv16, nW);
  wconv_kernel<<<nW / 1024, 256, 0, stream>>>(Wo, Wo16, nW);

  dim3 gg(BSZ * SEQ / 128, DMODEL / 64), gb(256);
  qkv_proj_kernel<<<gg, gb, 0, stream>>>(q, Wq16, bq, Qh);
  qkv_proj_kernel<<<gg, gb, 0, stream>>>(k, Wk16, bk, Kh);
  qkv_proj_kernel<<<gg, gb, 0, stream>>>(v, Wv16, bv, Vh);

  attn_kernel<<<dim3(SEQ / 64, BSZ * NHEAD), 128, 0, stream>>>(Qh, Kh, Vh, mask, Yc);

  out_proj_kernel<<<gg, gb, 0, stream>>>(Yc, Wo16, bo, out);
}
